// CWT_36653250904142
// MI455X (gfx1250) — matrix operand order repaired and checked by reading, not compiled, not run on hardware
//
#include <hip/hip_runtime.h>
#include <stdint.h>

// ---------------------------------------------------------------------------
// CWT (pywt 'morl') on MI455X/gfx1250 via v_wmma_f32_16x16x32_f16.
//
// out[n,t,c,s] = -sqrt(s) * sum_k g_s[k] * x[n, t - 8s - 1 + k, c]
//   g_s[k] = f_s[k-1] - f_s[k],  f_s[i] = int_psi[(i*1023)/(16*s)], L=16s+1
//
// GEMM mapping per (16t x 16c) tile:  D = A x B
//   A[m,K] = -sqrt(s)*g_s[K - m]   (Toeplitz; precomputed per (scale, K-chunk,
//                                   wave-phase) in exact ISA A-fragment layout)
//   B[K,n] = x16[t0w + a_s + K, c0+n]  (staged in LDS, async double-buffered)
// ---------------------------------------------------------------------------

typedef __attribute__((ext_vector_type(16))) _Float16 v16h;
typedef __attribute__((ext_vector_type(8)))  float    v8f;

#define NB      16
#define TLEN    2048
#define NCH     64
#define NSCALE  5
#define PADLO   2720          // 8*336 + 1, rounded up to 32
#define TPAD    7488          // 2048 + 2*2720
#define NCHUNK  170           // K chunks of 32 covering [t0-2720, t0+2688+31]
#define NPSI    1024

__constant__ int c_W[NSCALE]  = {1, 27, 76, 167, 336};
__constant__ int c_LG[NSCALE] = {18, 434, 1218, 2674, 5378}; // 16s+2

// ---------------------------------------------------------------------------
// Kernel 1: int_psi = cumsum(exp(-t^2/2)*cos(5t)) * step on 1024 points.
// ---------------------------------------------------------------------------
__global__ __launch_bounds__(1024) void k_intpsi(float* __restrict__ int_psi) {
  __shared__ float s[NPSI];
  const int i = threadIdx.x;
  const float step = 16.0f / 1023.0f;
  float t = -8.0f + (float)i * step;
  s[i] = expf(-0.5f * t * t) * cosf(5.0f * t);
  __syncthreads();
  for (int off = 1; off < NPSI; off <<= 1) {   // Hillis-Steele inclusive scan
    float v = (i >= off) ? s[i - off] : 0.0f;
    __syncthreads();
    s[i] += v;
    __syncthreads();
  }
  int_psi[i] = s[i] * step;
}

// ---------------------------------------------------------------------------
// Kernel 2: convert x (n,t,c) f32 -> zero-padded f16 xpad[n][c][TPAD].
// Channel-major so the hot loop's 32t x 64c chunk is 64 rows of 64 contiguous
// bytes -> one 16B async-to-LDS transfer per lane.
// ---------------------------------------------------------------------------
__global__ __launch_bounds__(256) void k_xpad(const float* __restrict__ x,
                                              _Float16* __restrict__ xpad) {
  size_t idx = (size_t)blockIdx.x * 256 + threadIdx.x;
  const size_t total = (size_t)NB * NCH * TPAD;
  if (idx >= total) return;
  int tp = (int)(idx % TPAD);
  int c  = (int)((idx / TPAD) % NCH);
  int n  = (int)(idx / ((size_t)TPAD * NCH));
  int t  = tp - PADLO;
  float v = 0.0f;
  if (t >= 0 && t < TLEN) v = x[((size_t)n * TLEN + t) * NCH + c];
  xpad[idx] = (_Float16)v;
}

// ---------------------------------------------------------------------------
// Kernel 3: A-fragment table in exact gfx1250 16-bit 16x32 A layout.
//   lanes 0-15: M=lane, halves h: v0-3 -> K=8*kh+h (h<8), v4-7 -> K=16+8*kh+h-8
//   (kh = lane>>4). value = -sqrt(s) * g_s[koff + K - m], koff per (s,j,dp).
// Table is t0-independent -> shared by every workgroup, lives in L2 (1.7MB).
// Entries outside a scale's active K-window are zero, so loads for any
// j in [0,NCHUNK) are always in-bounds and safe.
// ---------------------------------------------------------------------------
__device__ __forceinline__ float fval(int i, int sv, const float* ip) {
  if (i < 0 || i > 16 * sv) return 0.0f;        // f has L = 16s+1 taps
  return ip[(i * 1023) / (16 * sv)];
}

__global__ __launch_bounds__(256) void k_atab(const float* __restrict__ int_psi,
                                              _Float16* __restrict__ atab) {
  size_t idx = (size_t)blockIdx.x * 256 + threadIdx.x;
  const size_t total = (size_t)NSCALE * NCHUNK * 2 * 512;
  if (idx >= total) return;
  int h    = (int)(idx & 15);
  int lane = (int)((idx >> 4) & 31);
  int dp   = (int)((idx >> 9) & 1);
  int j    = (int)((idx >> 10) % NCHUNK);
  int s    = (int)(idx / ((size_t)NCHUNK * 1024));
  int sv = c_W[s], Lg = c_LG[s];
  int m  = lane & 15, kh = lane >> 4;
  int K  = (h < 8) ? (8 * kh + h) : (16 + 8 * kh + (h - 8));
  int koff = 32 * j - PADLO + 8 * sv + 1 - 16 * dp;
  int gi = koff + K - m;
  float v = 0.0f;
  if (gi >= 0 && gi < Lg) {
    float g = fval(gi - 1, sv, int_psi) - fval(gi, sv, int_psi);
    v = -sqrtf((float)sv) * g;
  }
  atab[idx] = (_Float16)v;
}

// ---------------------------------------------------------------------------
// Kernel 4: main WMMA kernel.
// Grid (64, 16): block = (t0 = 32 times) x (n). 256 threads = 8 waves:
//   wave -> (dtp = wave>>2 in {0,1} -> t_tile, ctile = (wave&3)*16).
// Per chunk j: async-stage 32t x 64c f16 into LDS (double buffer).
// Scale activity is a scalar range test (jlo/jhi precomputed + readfirstlane)
// so the hot loop predicate is SALU-only: no VALU chain, no WMMA->VALU NOPs.
// Loads (2x ds_load_b128 B + up to 5x 2 global_load_b128 A) are issued in one
// pass, then all WMMAs in a second pass -> single wait per chunk.
// ---------------------------------------------------------------------------
__device__ __forceinline__ void async_chunk16(uint32_t lds_addr, const _Float16* g) {
  unsigned long long ga = (unsigned long long)(uintptr_t)g;
  asm volatile("global_load_async_to_lds_b128 %0, %1, off"
               :: "v"(lds_addr), "v"(ga) : "memory");
}

__device__ __forceinline__ v16h load_frag32B(const void* p) {
  v16h r;
  __builtin_memcpy(&r, p, 32);   // 2x b128 loads (ds or global by addrspace)
  return r;
}

__global__ __launch_bounds__(256) void k_cwt(const _Float16* __restrict__ xpad,
                                             const _Float16* __restrict__ atab,
                                             float* __restrict__ out) {
  // 80B-padded rows: bank stride 20 dwords -> conflict-free 16-lane b128 reads
  __shared__ __align__(16) _Float16 chunk[2][NCH][40];

  const int t0   = blockIdx.x * 32;
  const int n    = blockIdx.y;
  const int tid  = threadIdx.x;
  const int wave = tid >> 5, lane = tid & 31;
  const int dtp  = wave >> 2;               // wave-phase: t_tile = t0 + 16*dtp
  const int t0w  = t0 + 16 * dtp;
  const int ctile = (wave & 3) * 16;
  const int ln = lane & 15, kh = lane >> 4;

  // async staging assignment: thread -> (channel, 8-time segment)
  const int cs   = tid & 63;
  const int tseg = tid >> 6;                // 0..3
  const _Float16* xn = xpad + (size_t)(n * NCH + cs) * TPAD + tseg * 8;
  const uint32_t ldsbase[2] = {
      (uint32_t)(uintptr_t)&chunk[0][cs][tseg * 8],
      (uint32_t)(uintptr_t)&chunk[1][cs][tseg * 8]};

  // Per-scale active chunk range: koff = 32j - (2719 - 8s + 16dtp) must lie in
  // (-32, LG+15).  C >= 0 always, so jlo = C/32, jhi = (C+LG+14)/32 + 1.
  // Scalarize once -> in-loop test is s_cmp on the uniform loop counter.
  int jlo[NSCALE], jhi[NSCALE];
  const _Float16* abase[NSCALE];
#pragma unroll
  for (int s = 0; s < NSCALE; ++s) {
    const int C = 2719 - 8 * c_W[s] + 16 * dtp;
    int lo = C >> 5;
    int hi = (C + c_LG[s] + 14) / 32 + 1;
    if (hi > NCHUNK) hi = NCHUNK;
    jlo[s] = __builtin_amdgcn_readfirstlane(lo);
    jhi[s] = __builtin_amdgcn_readfirstlane(hi);
    abase[s] = atab + ((((size_t)s * NCHUNK) * 2 + dtp) << 9) + (lane << 4);
  }

  v8f acc[NSCALE] = {};

  // prologue: chunk 0 (global time u = t0-2720+32j  ->  tp = t0 + 32j)
  async_chunk16(ldsbase[0], xn + t0);

  for (int j = 0; j < NCHUNK; ++j) {
    if (j + 1 < NCHUNK) {
      async_chunk16(ldsbase[(j + 1) & 1], xn + t0 + 32 * (j + 1));
      asm volatile("s_wait_asynccnt 1" ::: "memory");
    } else {
      asm volatile("s_wait_asynccnt 0" ::: "memory");
    }
    __syncthreads();   // chunk j visible to all waves

    // pass 1: issue all loads (A fragments from L2, B fragment from LDS)
    v16h af[NSCALE];
    bool act[NSCALE];
#pragma unroll
    for (int s = 0; s < NSCALE; ++s) {
      act[s] = (j >= jlo[s]) && (j < jhi[s]);          // SALU-only predicate
      if (act[s]) af[s] = load_frag32B(abase[s] + ((size_t)j << 10));
    }
    // B fragment: lane -> (chan = ctile+ln, K = 8*kh..8*kh+7, 16+8*kh..16+8*kh+7)
    v16h bf;
    __builtin_memcpy(&bf, &chunk[j & 1][ctile + ln][kh * 8], 16);
    __builtin_memcpy((char*)&bf + 16, &chunk[j & 1][ctile + ln][16 + kh * 8], 16);

    // pass 2: matrix ops back-to-back (distinct accumulators -> no hazards)
#pragma unroll
    for (int s = 0; s < NSCALE; ++s) {
      if (act[s]) {
        acc[s] = __builtin_amdgcn_wmma_f32_16x16x32_f16(
            false, af[s], false, bf, (short)0, acc[s], false, false);
      }
    }
    __syncthreads();   // all waves done with buf (j&1) before it is re-filled
  }

  // store: C/D layout -> t = t0w + r + 8*kh, c = ctile + ln; out[n][t][c][s]
#pragma unroll
  for (int r = 0; r < 8; ++r) {
    const int t = t0w + r + 8 * kh;
    const size_t o = (((size_t)n * TLEN + t) * NCH + ctile + ln) * NSCALE;
#pragma unroll
    for (int s = 0; s < NSCALE; ++s) out[o + s] = acc[s][r];
  }
}

// ---------------------------------------------------------------------------
// Launch: ws layout (bytes):
//   [0,4K)        int_psi f32[1024]
//   [8K, ~1.75M)  A-fragment table f16 (5*170*2*512)
//   [2M, ~17.4M)  xpad f16 [16][64][7488]
// ---------------------------------------------------------------------------
extern "C" void kernel_launch(void* const* d_in, const int* in_sizes, int n_in,
                              void* d_out, int out_size, void* d_ws, size_t ws_size,
                              hipStream_t stream) {
  (void)in_sizes; (void)n_in; (void)out_size; (void)ws_size;
  const float* x = (const float*)d_in[0];
  float* out = (float*)d_out;
  char* ws = (char*)d_ws;

  float*    int_psi = (float*)ws;
  _Float16* atab    = (_Float16*)(ws + 8192);
  _Float16* xpad    = (_Float16*)(ws + (size_t)(2u << 20));

  k_intpsi<<<1, 1024, 0, stream>>>(int_psi);

  const size_t xtot = (size_t)NB * NCH * TPAD;
  k_xpad<<<(unsigned)((xtot + 255) / 256), 256, 0, stream>>>(x, xpad);

  const size_t atot = (size_t)NSCALE * NCHUNK * 2 * 512;
  k_atab<<<(unsigned)((atot + 255) / 256), 256, 0, stream>>>(int_psi, atab);

  dim3 grid(TLEN / 32, NB);
  k_cwt<<<grid, 256, 0, stream>>>(xpad, atab, out);
}
